// SpatialReductionAttention_58093727645843
// MI455X (gfx1250) — compile-verified, matrix operand order checked by reading, not run on hardware
//
#include <hip/hip_runtime.h>
#include <hip/hip_bf16.h>

// ---------------------------------------------------------------------------
// SpatialReductionAttention for MI455X (gfx1250), wave32 + WMMA bf16 16x16x32.
// All GEMM stages run on v_wmma_f32_16x16x32_bf16 with f32 accumulation.
// Attention K/V staging uses the Tensor Data Mover (tensor_load_to_lds).
// ---------------------------------------------------------------------------

typedef __attribute__((ext_vector_type(16))) __bf16 v16bf;
typedef __attribute__((ext_vector_type(8)))  __bf16 v8bf;
typedef __attribute__((ext_vector_type(4)))  __bf16 v4bf;
typedef __attribute__((ext_vector_type(8)))  float  v8f;
typedef __attribute__((ext_vector_type(4)))  unsigned int v4u;
typedef __attribute__((ext_vector_type(8)))  int v8i;
typedef __attribute__((ext_vector_type(4)))  int v4i;

#define NB_HEADS 8
#define HEAD_DIM 32
#define CDIM     256
#define NTOK     4096   // 64*64
#define BATCH    16
#define MTOK     256    // 16*16 after SR=4

// Native bf16 conversions (lower to v_cvt_pk_bf16_f32)
__device__ __forceinline__ void st_bf16(unsigned short* p, float f) {
  *reinterpret_cast<__bf16*>(p) = (__bf16)f;
}
__device__ __forceinline__ void st_bf16x4(unsigned short* p, float4 v) {
  v4bf r;
  r[0] = (__bf16)v.x; r[1] = (__bf16)v.y; r[2] = (__bf16)v.z; r[3] = (__bf16)v.w;
  *reinterpret_cast<v4bf*>(p) = r;
}
__device__ __forceinline__ unsigned pack2_bf16(float a, float b) {
  union { __bf16 h[2]; unsigned u; } r;
  r.h[0] = (__bf16)a; r.h[1] = (__bf16)b;
  return r.u;
}

// A/B fragment for V_WMMA_*_16x16x32 (16-bit): lane l holds row (l&15),
// K = {0..7, 16..23} for lanes 0-15, {8..15, 24..31} for lanes 16-31.
// -> two contiguous 16B loads from a bf16 row of 32 elements.
__device__ __forceinline__ v16bf frag_load(const unsigned short* row, int hi) {
  const __bf16* p = reinterpret_cast<const __bf16*>(row) + hi * 8;
  v8bf lo = *reinterpret_cast<const v8bf*>(p);
  v8bf hh = *reinterpret_cast<const v8bf*>(p + 16);
  v16bf r;
#pragma unroll
  for (int i = 0; i < 8; ++i) { r[i] = lo[i]; r[8 + i] = hh[i]; }
  return r;
}

__device__ __forceinline__ v8f wmma_bf16(v16bf a, v16bf b, v8f c) {
  return __builtin_amdgcn_wmma_f32_16x16x32_bf16(
      false, a, false, b, (short)0, c, false, false);
}

__device__ __forceinline__ void zero8(v8f& v) {
#pragma unroll
  for (int r = 0; r < 8; ++r) v[r] = 0.0f;
}

// Tensor Data Mover: 1-D contiguous copy of `nelem` 2-byte elements from
// global memory to LDS byte offset `lds_off`. Builds a D# per CDNA5 ISA:
// group0 = {count=1 | type=2, lds_addr, global_addr[56:0]},
// group1 = {data_size=2B, tensor_dim0 = tile_dim0 = nelem, dim1=1}.
// This toolchain exposes the 6-arg builtin (g0, g1, g2, g3, g4, cpol).
__device__ __forceinline__ void tdm_load_1d(unsigned lds_off, const void* gptr,
                                            unsigned nelem) {
  unsigned long long ga = (unsigned long long)gptr;
  v4u g0;
  g0[0] = 1u;                                            // count=1, user mode
  g0[1] = lds_off;                                       // lds_addr (bytes)
  g0[2] = (unsigned)(ga & 0xFFFFFFFFu);                  // global_addr lo
  g0[3] = (unsigned)((ga >> 32) & 0x01FFFFFFu) | (2u << 30); // addr hi | type=2
  v8i g1;
  g1[0] = (int)(1u << 16);                               // data_size=1 (2B)
  g1[1] = (int)((nelem & 0xFFFFu) << 16);                // tensor_dim0 lo16
  g1[2] = (int)((nelem >> 16) & 0xFFFFu) | (1 << 16);    // dim0 hi | dim1=1
  g1[3] = (int)((nelem & 0xFFFFu) << 16);                // tile_dim0
  g1[4] = 0;                                             // tile_dim1/2 = 0
  g1[5] = (int)nelem;                                    // dim0 stride
  g1[6] = 0; g1[7] = 0;
  v4i z4; z4[0] = 0; z4[1] = 0; z4[2] = 0; z4[3] = 0;
  v8i z8;
#pragma unroll
  for (int i = 0; i < 8; ++i) z8[i] = 0;
  __builtin_amdgcn_tensor_load_to_lds(g0, g1, z4, z4, z8, 0);
}

// ---------------------------------------------------------------------------
// Kernel 1: Q projection. x[65536,256] * q_w[256,256] + q_b
// Output bf16 in per-head layout q[b][h][n][hd].
// ---------------------------------------------------------------------------
__global__ __launch_bounds__(128) void k_qproj(
    const float* __restrict__ x, const float* __restrict__ qw,
    const float* __restrict__ qb, unsigned short* __restrict__ qout) {
  __shared__ unsigned short sA[16 * 32];
  __shared__ unsigned short sBT[256 * 32];   // B transposed: [n][k] bf16
  unsigned* sBT32 = reinterpret_cast<unsigned*>(sBT);
  const int tid = threadIdx.x, lane = tid & 31, wv = tid >> 5;
  const int hi = lane >> 4, ln = lane & 15;
  const int row0 = blockIdx.x * 16;

  v8f acc[4];
#pragma unroll
  for (int t = 0; t < 4; ++t) zero8(acc[t]);

  for (int kk0 = 0; kk0 < CDIM; kk0 += 32) {
    { // stage A (16x32 fp32 -> bf16): 128 thr x 1 float4
      int r = tid >> 3, c4 = (tid & 7) * 4;
      float4 v4 = *reinterpret_cast<const float4*>(
          x + (size_t)(row0 + r) * CDIM + kk0 + c4);
      st_bf16x4(sA + r * 32 + c4, v4);
    }
#pragma unroll
    for (int it = 0; it < 8; ++it) {  // stage B^T in K-row pairs (packed cvt)
      int idx = tid + it * 128;       // 0..1023
      int kp = idx >> 6;              // K row pair 0..15
      int n  = (idx & 63) * 4;        // column 0..252
      float4 w0 = *reinterpret_cast<const float4*>(
          qw + (size_t)(kk0 + 2 * kp) * CDIM + n);
      float4 w1 = *reinterpret_cast<const float4*>(
          qw + (size_t)(kk0 + 2 * kp + 1) * CDIM + n);
      sBT32[(n + 0) * 16 + kp] = pack2_bf16(w0.x, w1.x);
      sBT32[(n + 1) * 16 + kp] = pack2_bf16(w0.y, w1.y);
      sBT32[(n + 2) * 16 + kp] = pack2_bf16(w0.z, w1.z);
      sBT32[(n + 3) * 16 + kp] = pack2_bf16(w0.w, w1.w);
    }
    if (kk0 + 32 < CDIM)
      __builtin_prefetch(qw + (size_t)(kk0 + 32) * CDIM + tid * 8, 0, 0);
    __syncthreads();
    v16bf a = frag_load(sA + ln * 32, hi);
#pragma unroll
    for (int t = 0; t < 4; ++t) {
      int ct = wv * 4 + t;
      v16bf b = frag_load(sBT + (ct * 16 + ln) * 32, hi);
      acc[t] = wmma_bf16(a, b, acc[t]);
    }
    __syncthreads();
  }
#pragma unroll
  for (int t = 0; t < 4; ++t) {
    int col = (wv * 4 + t) * 16 + ln;
    int head = col >> 5, within = col & 31;
    float bias = qb[col];
#pragma unroll
    for (int r = 0; r < 8; ++r) {
      int gr = row0 + r + 8 * hi;
      int bidx = gr >> 12, n = gr & (NTOK - 1);
      st_bf16(qout + (size_t)((bidx * NB_HEADS + head) * NTOK + n) * HEAD_DIM +
                  within,
              acc[t][r] + bias);
    }
  }
}

// ---------------------------------------------------------------------------
// Kernel 2: spatial-reduction conv (k=s=4 => patch GEMM, K=4096) + bias + LN.
// Rows = B*M = 4096. Block: 256 thr (8 waves), 32 rows x 256 cols.
// ---------------------------------------------------------------------------
__global__ __launch_bounds__(256) void k_srconv_ln(
    const float* __restrict__ x, const float* __restrict__ srw,
    const float* __restrict__ srb, const float* __restrict__ ng,
    const float* __restrict__ nb, unsigned short* __restrict__ xsout) {
  __shared__ unsigned short sA[32 * 32];
  __shared__ unsigned short sBT[256 * 32];
  __shared__ float sXS[32][264];
  unsigned* sBT32 = reinterpret_cast<unsigned*>(sBT);
  const int tid = threadIdx.x, lane = tid & 31, wv = tid >> 5;
  const int hi = lane >> 4, ln = lane & 15;
  const int row0 = blockIdx.x * 32;       // xs row
  const int rt = wv >> 2;                 // row tile 0..1
  const int ctg = (wv & 3) * 4;           // first col tile of this wave

  v8f acc[4];
#pragma unroll
  for (int t = 0; t < 4; ++t) zero8(acc[t]);

  for (int kk0 = 0; kk0 < 4096; kk0 += 32) {
    int ij = kk0 >> 8, cin0 = kk0 & 255;
    int ii = ij >> 2, jj = ij & 3;
    { // gather-stage A: 32 rows x 32 (one 4x4-patch channel chunk)
      int r = tid >> 3, c4 = (tid & 7) * 4;
      int gr = row0 + r;
      int b = gr >> 8, m = gr & 255;
      int mh = m >> 4, mw = m & 15;
      float4 v4 = *reinterpret_cast<const float4*>(
          x + (size_t)((b * 64 + mh * 4 + ii) * 64 + (mw * 4 + jj)) * CDIM +
          cin0 + c4);
      st_bf16x4(sA + r * 32 + c4, v4);
    }
#pragma unroll
    for (int it = 0; it < 4; ++it) {  // stage B^T in K-row pairs (packed cvt)
      int idx = tid + it * 256;       // 0..1023
      int kp = idx >> 6;              // 0..15
      int n  = (idx & 63) * 4;
      float4 w0 = *reinterpret_cast<const float4*>(
          srw + (size_t)(kk0 + 2 * kp) * CDIM + n);
      float4 w1 = *reinterpret_cast<const float4*>(
          srw + (size_t)(kk0 + 2 * kp + 1) * CDIM + n);
      sBT32[(n + 0) * 16 + kp] = pack2_bf16(w0.x, w1.x);
      sBT32[(n + 1) * 16 + kp] = pack2_bf16(w0.y, w1.y);
      sBT32[(n + 2) * 16 + kp] = pack2_bf16(w0.z, w1.z);
      sBT32[(n + 3) * 16 + kp] = pack2_bf16(w0.w, w1.w);
    }
    if (kk0 + 32 < 4096)
      __builtin_prefetch(srw + (size_t)(kk0 + 32) * CDIM + tid * 8, 0, 0);
    __syncthreads();
    v16bf a = frag_load(sA + (rt * 16 + ln) * 32, hi);
#pragma unroll
    for (int t = 0; t < 4; ++t) {
      v16bf b = frag_load(sBT + ((ctg + t) * 16 + ln) * 32, hi);
      acc[t] = wmma_bf16(a, b, acc[t]);
    }
    __syncthreads();
  }
  // bias + stash fp32 block for LayerNorm
#pragma unroll
  for (int t = 0; t < 4; ++t) {
    int col = (ctg + t) * 16 + ln;
    float bias = srb[col];
#pragma unroll
    for (int r = 0; r < 8; ++r)
      sXS[rt * 16 + r + 8 * hi][col] = acc[t][r] + bias;
  }
  __syncthreads();
  { // LayerNorm over C=256, 8 threads per row
    int r = tid >> 3, seg = tid & 7;
    float s = 0.f, s2 = 0.f;
#pragma unroll
    for (int c = 0; c < 32; ++c) {
      float v = sXS[r][seg * 32 + c];
      s += v; s2 += v * v;
    }
#pragma unroll
    for (int msk = 1; msk < 8; msk <<= 1) {
      s  += __shfl_xor(s,  msk, 32);
      s2 += __shfl_xor(s2, msk, 32);
    }
    float mu  = s  * (1.0f / 256.0f);
    float var = s2 * (1.0f / 256.0f) - mu * mu;
    float inv = rsqrtf(var + 1e-6f);
    int gr = row0 + r;
#pragma unroll
    for (int c8 = 0; c8 < 8; ++c8) {  // packed 4-wide bf16 stores
      int col = seg * 32 + c8 * 4;
      float4 o;
      o.x = (sXS[r][col + 0] - mu) * inv * ng[col + 0] + nb[col + 0];
      o.y = (sXS[r][col + 1] - mu) * inv * ng[col + 1] + nb[col + 1];
      o.z = (sXS[r][col + 2] - mu) * inv * ng[col + 2] + nb[col + 2];
      o.w = (sXS[r][col + 3] - mu) * inv * ng[col + 3] + nb[col + 3];
      st_bf16x4(xsout + (size_t)gr * CDIM + col, o);
    }
  }
}

// ---------------------------------------------------------------------------
// Kernel 3: KV projection. xs_bf16[4096,256] * kv_w[256,512] + kv_b.
// Output k,v bf16 in [b][h][m][hd] layout.
// ---------------------------------------------------------------------------
__global__ __launch_bounds__(128) void k_kv(
    const unsigned short* __restrict__ xs, const float* __restrict__ kvw,
    const float* __restrict__ kvb, unsigned short* __restrict__ kout,
    unsigned short* __restrict__ vout) {
  __shared__ unsigned short sA[16 * 32];
  __shared__ unsigned short sBT[512 * 32];
  unsigned* sBT32 = reinterpret_cast<unsigned*>(sBT);
  const int tid = threadIdx.x, lane = tid & 31, wv = tid >> 5;
  const int hi = lane >> 4, ln = lane & 15;
  const int row0 = blockIdx.x * 16;

  v8f acc[8];
#pragma unroll
  for (int t = 0; t < 8; ++t) zero8(acc[t]);

  for (int kk0 = 0; kk0 < CDIM; kk0 += 32) {
    if (tid < 64) {  // stage A: straight bf16 copy, 16x32
      int r = tid >> 2, c8 = (tid & 3) * 8;
      *reinterpret_cast<float4*>(sA + r * 32 + c8) =
          *reinterpret_cast<const float4*>(xs + (size_t)(row0 + r) * CDIM +
                                           kk0 + c8);
    }
#pragma unroll
    for (int it = 0; it < 16; ++it) {  // stage B^T (32x512) in K-row pairs
      int idx = tid + it * 128;        // 0..2047
      int kp = idx >> 7;               // 0..15
      int n  = (idx & 127) * 4;        // 0..508
      float4 w0 = *reinterpret_cast<const float4*>(
          kvw + (size_t)(kk0 + 2 * kp) * 512 + n);
      float4 w1 = *reinterpret_cast<const float4*>(
          kvw + (size_t)(kk0 + 2 * kp + 1) * 512 + n);
      sBT32[(n + 0) * 16 + kp] = pack2_bf16(w0.x, w1.x);
      sBT32[(n + 1) * 16 + kp] = pack2_bf16(w0.y, w1.y);
      sBT32[(n + 2) * 16 + kp] = pack2_bf16(w0.z, w1.z);
      sBT32[(n + 3) * 16 + kp] = pack2_bf16(w0.w, w1.w);
    }
    __syncthreads();
    v16bf a = frag_load(sA + ln * 32, hi);
#pragma unroll
    for (int t = 0; t < 8; ++t) {
      int ct = wv * 8 + t;
      v16bf b = frag_load(sBT + (ct * 16 + ln) * 32, hi);
      acc[t] = wmma_bf16(a, b, acc[t]);
    }
    __syncthreads();
  }
#pragma unroll
  for (int t = 0; t < 8; ++t) {
    int col = (wv * 8 + t) * 16 + ln;
    int kvsel = col >> 8, head = (col >> 5) & 7, within = col & 31;
    float bias = kvb[col];
    unsigned short* dst = kvsel ? vout : kout;
#pragma unroll
    for (int r = 0; r < 8; ++r) {
      int gr = row0 + r + 8 * hi;
      int b = gr >> 8, m = gr & 255;
      st_bf16(dst + (size_t)((b * NB_HEADS + head) * MTOK + m) * HEAD_DIM +
                  within,
              acc[t][r] + bias);
    }
  }
}

// ---------------------------------------------------------------------------
// Kernel 4: fused attention per (b,h): S = scale*q@k^T, softmax, O = P@v.
// Block: 128 thr (4 waves), each wave owns 16 query rows (full M=256 context).
// K and raw V are brought into LDS by the Tensor Data Mover (async DMA,
// TENSORcnt), V is transposed LDS->LDS, probabilities go C-layout -> A-layout
// through per-wave LDS buffers.
// LDS map (bytes): sK @ 0 (16K) | sVT @ 16384 (16K) | sP @ 32768 (32K).
// ---------------------------------------------------------------------------
__global__ __launch_bounds__(128) void k_attn(
    const unsigned short* __restrict__ q, const unsigned short* __restrict__ kk,
    const unsigned short* __restrict__ vv, unsigned short* __restrict__ out) {
  __shared__ unsigned short smem[2 * MTOK * HEAD_DIM + 4 * 16 * MTOK];
  unsigned short* sK  = smem;                        // [m][d]
  unsigned short* sVT = smem + MTOK * HEAD_DIM;      // [d][m]
  unsigned short* sP  = smem + 2 * MTOK * HEAD_DIM;  // probs / raw-V staging
  const int tid = threadIdx.x, lane = tid & 31, wv = tid >> 5;
  const int hi = lane >> 4, ln = lane & 15;
  const int gid = blockIdx.x;
  const int bh = gid >> 6, nblk = gid & 63;
  const int b = bh >> 3, h = bh & 7;
  const unsigned short* kbase = kk + (size_t)(b * NB_HEADS + h) * MTOK * HEAD_DIM;
  const unsigned short* vbase = vv + (size_t)(b * NB_HEADS + h) * MTOK * HEAD_DIM;
  const unsigned short* qbase = q + (size_t)(b * NB_HEADS + h) * NTOK * HEAD_DIM;

  if (wv == 0) {  // TDM: K -> LDS@0, raw V -> LDS@32768 (16KB each)
    tdm_load_1d(0u, kbase, MTOK * HEAD_DIM);
    tdm_load_1d(2u * MTOK * HEAD_DIM * 2u, vbase, MTOK * HEAD_DIM);
    __builtin_amdgcn_s_wait_tensorcnt(0);
  }
  __syncthreads();
#pragma unroll
  for (int it = 0; it < 64; ++it) {  // transpose V: sP raw [m][d] -> sVT [d][m]
    int idx = tid + it * 128;
    int m = idx >> 5, c = idx & 31;
    sVT[c * MTOK + m] = sP[m * HEAD_DIM + c];
  }
  __syncthreads();

  const int n0 = nblk * 64 + wv * 16;
  v16bf aq = frag_load(qbase + (size_t)(n0 + ln) * HEAD_DIM, hi);

  v8f s[16];
#pragma unroll
  for (int ct = 0; ct < 16; ++ct) zero8(s[ct]);
#pragma unroll
  for (int ct = 0; ct < 16; ++ct) {  // S tile: hd=32 fits one WMMA K
    v16bf bk = frag_load(sK + (ct * 16 + ln) * HEAD_DIM, hi);
    s[ct] = wmma_bf16(aq, bk, s[ct]);
  }

  const float scale = 0.17677669529663689f;  // 1/sqrt(32)
  unsigned short* pbuf = sP + wv * 16 * MTOK;
#pragma unroll
  for (int r = 0; r < 8; ++r) {  // softmax over M=256 per query row
    float mx = -3.402823466e38f;
#pragma unroll
    for (int ct = 0; ct < 16; ++ct) mx = fmaxf(mx, s[ct][r]);
#pragma unroll
    for (int msk = 1; msk < 16; msk <<= 1)
      mx = fmaxf(mx, __shfl_xor(mx, msk, 32));
    float sum = 0.f;
#pragma unroll
    for (int ct = 0; ct < 16; ++ct) {
      float e = __expf((s[ct][r] - mx) * scale);
      s[ct][r] = e; sum += e;
    }
#pragma unroll
    for (int msk = 1; msk < 16; msk <<= 1) sum += __shfl_xor(sum, msk, 32);
    float inv = 1.0f / sum;
    int prow = r + 8 * hi;
#pragma unroll
    for (int ct = 0; ct < 16; ++ct)
      st_bf16(pbuf + prow * MTOK + ct * 16 + ln, s[ct][r] * inv);
  }
  asm volatile("s_wait_dscnt 0" ::: "memory");  // wave-local LDS RAW fence

  v8f o[2];
  zero8(o[0]); zero8(o[1]);
#pragma unroll
  for (int c = 0; c < 8; ++c) {  // O = P[16x256] @ V[256x32]
    v16bf ap = frag_load(pbuf + ln * MTOK + c * 32, hi);
#pragma unroll
    for (int t = 0; t < 2; ++t) {
      v16bf bv = frag_load(sVT + (t * 16 + ln) * MTOK + c * 32, hi);
      o[t] = wmma_bf16(ap, bv, o[t]);
    }
  }
#pragma unroll
  for (int t = 0; t < 2; ++t)
#pragma unroll
    for (int r = 0; r < 8; ++r) {
      int n = n0 + r + 8 * hi;
      st_bf16(out + (size_t)(b * NTOK + n) * CDIM + h * HEAD_DIM + t * 16 + ln,
              o[t][r]);
    }
}

// ---------------------------------------------------------------------------
// Kernel 5: output projection. ao_bf16[65536,256] * proj_w[256,256] + proj_b
// -> fp32 d_out.
// ---------------------------------------------------------------------------
__global__ __launch_bounds__(128) void k_proj(
    const unsigned short* __restrict__ ao, const float* __restrict__ pw,
    const float* __restrict__ pb, float* __restrict__ outp) {
  __shared__ unsigned short sA[16 * 32];
  __shared__ unsigned short sBT[256 * 32];
  unsigned* sBT32 = reinterpret_cast<unsigned*>(sBT);
  const int tid = threadIdx.x, lane = tid & 31, wv = tid >> 5;
  const int hi = lane >> 4, ln = lane & 15;
  const int row0 = blockIdx.x * 16;

  v8f acc[4];
#pragma unroll
  for (int t = 0; t < 4; ++t) zero8(acc[t]);

  for (int kk0 = 0; kk0 < CDIM; kk0 += 32) {
    if (tid < 64) {  // A already bf16: straight copy
      int r = tid >> 2, c8 = (tid & 3) * 8;
      *reinterpret_cast<float4*>(sA + r * 32 + c8) =
          *reinterpret_cast<const float4*>(ao + (size_t)(row0 + r) * CDIM +
                                           kk0 + c8);
    }
#pragma unroll
    for (int it = 0; it < 8; ++it) {  // stage B^T in K-row pairs (packed cvt)
      int idx = tid + it * 128;       // 0..1023
      int kp = idx >> 6;              // 0..15
      int n  = (idx & 63) * 4;
      float4 w0 = *reinterpret_cast<const float4*>(
          pw + (size_t)(kk0 + 2 * kp) * CDIM + n);
      float4 w1 = *reinterpret_cast<const float4*>(
          pw + (size_t)(kk0 + 2 * kp + 1) * CDIM + n);
      sBT32[(n + 0) * 16 + kp] = pack2_bf16(w0.x, w1.x);
      sBT32[(n + 1) * 16 + kp] = pack2_bf16(w0.y, w1.y);
      sBT32[(n + 2) * 16 + kp] = pack2_bf16(w0.z, w1.z);
      sBT32[(n + 3) * 16 + kp] = pack2_bf16(w0.w, w1.w);
    }
    __syncthreads();
    v16bf a = frag_load(sA + ln * 32, hi);
#pragma unroll
    for (int t = 0; t < 4; ++t) {
      int ct = wv * 4 + t;
      v16bf b = frag_load(sBT + (ct * 16 + ln) * 32, hi);
      acc[t] = wmma_bf16(a, b, acc[t]);
    }
    __syncthreads();
  }
#pragma unroll
  for (int t = 0; t < 4; ++t) {
    int col = (wv * 4 + t) * 16 + ln;
    float bias = pb[col];
#pragma unroll
    for (int r = 0; r < 8; ++r) {
      int gr = row0 + r + 8 * hi;
      outp[(size_t)gr * CDIM + col] = acc[t][r] + bias;
    }
  }
}

// ---------------------------------------------------------------------------
extern "C" void kernel_launch(void* const* d_in, const int* in_sizes, int n_in,
                              void* d_out, int out_size, void* d_ws,
                              size_t ws_size, hipStream_t stream) {
  const float* x   = (const float*)d_in[0];
  const float* qw  = (const float*)d_in[1];
  const float* qb  = (const float*)d_in[2];
  const float* kvw = (const float*)d_in[3];
  const float* kvb = (const float*)d_in[4];
  const float* srw = (const float*)d_in[5];
  const float* srb = (const float*)d_in[6];
  const float* ng  = (const float*)d_in[7];
  const float* nb  = (const float*)d_in[8];
  const float* pw  = (const float*)d_in[9];
  const float* pb  = (const float*)d_in[10];

  char* ws = (char*)d_ws;
  unsigned short* qbf  = (unsigned short*)(ws + 0);          // 32 MB
  unsigned short* xsbf = (unsigned short*)(ws + 33554432);   //  2 MB
  unsigned short* kbf  = (unsigned short*)(ws + 35651584);   //  2 MB
  unsigned short* vbf  = (unsigned short*)(ws + 37748736);   //  2 MB
  unsigned short* aobf = (unsigned short*)(ws + 39845888);   // 32 MB

  k_qproj    <<<4096, 128, 0, stream>>>(x, qw, qb, qbf);
  k_srconv_ln<<< 128, 256, 0, stream>>>(x, srw, srb, ng, nb, xsbf);
  k_kv       <<< 256, 128, 0, stream>>>(xsbf, kvw, kvb, kbf, vbf);
  k_attn     <<<8192, 128, 0, stream>>>(qbf, kbf, vbf, aobf);
  k_proj     <<<4096, 128, 0, stream>>>(aobf, pw, pb, (float*)d_out);
}
